// GATLayer_59322088292864
// MI455X (gfx1250) — matrix operand order repaired and checked by reading, not compiled, not run on hardware
//
#include <hip/hip_runtime.h>
#include <hip/hip_bf16.h>
#include <stdint.h>

// ---------------------------------------------------------------------------
// GAT layer, fused for CDNA5 (gfx1250, wave32, WMMA bf16 16x16x32 + TDM)
// ---------------------------------------------------------------------------
typedef __attribute__((ext_vector_type(16))) __bf16        v16bf;
typedef __attribute__((ext_vector_type(8)))  float         v8f;
typedef __attribute__((ext_vector_type(8)))  unsigned int  v8u;
typedef __attribute__((ext_vector_type(4)))  unsigned int  u32x4;
typedef __attribute__((ext_vector_type(8)))  int           i32x8;
typedef __attribute__((ext_vector_type(4)))  int           i32x4;

#define BATCH 4
#define NN    2048
#define DM    768
#define BNTOT (BATCH * NN)
#define ALPHA 0.2f

__device__ __forceinline__ float lrelu(float x) { return x > 0.0f ? x : ALPHA * x; }

__device__ __forceinline__ unsigned short f32_to_bf16(float x) {
    unsigned int u = __builtin_bit_cast(unsigned int, x);
    unsigned int r = u + 0x7FFFu + ((u >> 16) & 1u);   // round-to-nearest-even
    return (unsigned short)(r >> 16);
}

// ISA 7.12.2: 16-bit A-matrix 16x32. lanes 0-15: K=0..7 (v0..3), K=16..23 (v4..7);
// lanes 16-31: +8.  Pair (k,k+1) per VGPR -> one aligned b32 LDS read each.
__device__ __forceinline__ int kk_a(int v, int half) {
    return (v < 4 ? 2 * v : 16 + 2 * (v - 4)) + 8 * half;
}
// B-matrix 32x16: same K order as A (lanes 0-15: K=0..7,16..23; lanes 16-31: +8).
__device__ __forceinline__ int kk_b(int v, int half) {
    return (v < 4 ? 2 * v : 16 + 2 * (v - 4)) + 8 * half;
}

// ---------------------------------------------------------------------------
// Tensor Data Mover: issue a 2D tile load (tile_dim0 x tile_dim1, 2-byte
// elements) from global memory into LDS.  D# per CDNA5 ISA ch.8.
// This toolchain exposes the 6-arg builtin (g0, g1, g2, g3, g4, cpol).
// ---------------------------------------------------------------------------
__device__ __forceinline__ void tdm_load_2d_b16(unsigned ldsOff,
                                                const void* gaddr,
                                                unsigned tensorD0, unsigned tensorD1,
                                                unsigned tileD0, unsigned tileD1,
                                                unsigned strideD0)
{
    unsigned long long ga = (unsigned long long)(uintptr_t)gaddr;
    u32x4 g0;
    g0[0] = 1u;                                        // count=1, user mode
    g0[1] = ldsOff;                                    // lds_addr
    g0[2] = (unsigned)(ga & 0xFFFFFFFFu);              // global_addr[31:0]
    g0[3] = (unsigned)((ga >> 32) & 0x01FFFFFFu)       // global_addr[56:32]
          | (2u << 30);                                // type=2 ("image")
    i32x8 g1;
    g1[0] = (int)(1u << 16);                           // data_size=1 (2 bytes)
    g1[1] = (int)((tensorD0 & 0xFFFFu) << 16);         // tensor_dim0[15:0]
    g1[2] = (int)(((tensorD0 >> 16) & 0xFFFFu)
          | ((tensorD1 & 0xFFFFu) << 16));             // dim0 hi | tensor_dim1 lo
    g1[3] = (int)(((tensorD1 >> 16) & 0xFFFFu)
          | ((tileD0 & 0xFFFFu) << 16));               // dim1 hi | tile_dim0
    g1[4] = (int)(tileD1 & 0xFFFFu);                   // tile_dim1 (tile_dim2=0)
    g1[5] = (int)strideD0;                             // tensor_dim0_stride lo
    g1[6] = 0;                                         // stride hi | dim1_stride lo
    g1[7] = 0;
    i32x4 z4 = (i32x4)(0);
    i32x8 z8 = (i32x8)(0);
    __builtin_amdgcn_tensor_load_to_lds(g0, g1, z4, z4, z8, 0);
}

// ===========================================================================
// K1: projection GEMM  Wh = h @ W  (bf16 WMMA, f32 accumulate)
// grid (64 Mtiles, 6 Ntiles, 2 sources), block 256 (8 waves)
// ===========================================================================
__global__ __launch_bounds__(256)
void k1_proj(const float* __restrict__ h0, const float* __restrict__ h1,
             const float* __restrict__ W,
             float* __restrict__ Wh0f, float* __restrict__ Wh1f,
             unsigned short* __restrict__ Wh1b)
{
    __shared__ unsigned short ALds[128][34];   // rows x k (bf16)
    __shared__ unsigned short WLds[128][34];   // n    x k (bf16, transposed)

    const int t        = threadIdx.x;
    const int lane     = t & 31;
    const int half     = lane >> 4;
    const int waveId   = t >> 5;
    const int mWave    = (waveId >> 2) * 64;
    const int nWave    = (waveId & 3) * 32;
    const int rowBase  = blockIdx.x * 128;
    const int nBase    = blockIdx.y * 128;
    const int src      = blockIdx.z;
    const float* hsrc  = (src == 0) ? h0 : h1;

    v8f acc[4][2];
    #pragma unroll
    for (int mt = 0; mt < 4; ++mt)
        #pragma unroll
        for (int nt = 0; nt < 2; ++nt)
            acc[mt][nt] = (v8f)(0.0f);

    for (int k0 = 0; k0 < DM; k0 += 32) {
        __syncthreads();
        #pragma unroll
        for (int q = 0; q < 16; ++q) {
            int idx = t + 256 * q;            // 0..4095
            int row = idx >> 5, kk = idx & 31;
            ALds[row][kk] = f32_to_bf16(hsrc[(rowBase + row) * DM + k0 + kk]);
        }
        #pragma unroll
        for (int q = 0; q < 16; ++q) {
            int idx = t + 256 * q;            // 0..4095
            int n = idx & 127, kk = idx >> 7;
            WLds[n][kk] = f32_to_bf16(W[(k0 + kk) * DM + nBase + n]);
        }
        __syncthreads();

        v8u rawA[4];
        #pragma unroll
        for (int mt = 0; mt < 4; ++mt) {
            int r = mWave + mt * 16 + (lane & 15);
            #pragma unroll
            for (int v = 0; v < 8; ++v)
                rawA[mt][v] = *(const unsigned int*)&ALds[r][kk_a(v, half)];
        }
        #pragma unroll
        for (int nt = 0; nt < 2; ++nt) {
            v8u rawB;
            int n = nWave + nt * 16 + (lane & 15);
            #pragma unroll
            for (int v = 0; v < 8; ++v)
                rawB[v] = *(const unsigned int*)&WLds[n][kk_b(v, half)];
            v16bf bF = __builtin_bit_cast(v16bf, rawB);
            #pragma unroll
            for (int mt = 0; mt < 4; ++mt) {
                v16bf aF = __builtin_bit_cast(v16bf, rawA[mt]);
                acc[mt][nt] = __builtin_amdgcn_wmma_f32_16x16x32_bf16(
                    false, aF, false, bF, (short)0, acc[mt][nt], false, false);
            }
        }
    }

    if (src == 0) {
        #pragma unroll
        for (int mt = 0; mt < 4; ++mt)
            #pragma unroll
            for (int nt = 0; nt < 2; ++nt)
                #pragma unroll
                for (int r = 0; r < 8; ++r) {
                    int gm = rowBase + mWave + mt * 16 + r + 8 * half;
                    int gn = nBase + nWave + nt * 16 + (lane & 15);
                    Wh0f[gm * DM + gn] = acc[mt][nt][r];
                }
    } else {
        #pragma unroll
        for (int mt = 0; mt < 4; ++mt)
            #pragma unroll
            for (int nt = 0; nt < 2; ++nt)
                #pragma unroll
                for (int r = 0; r < 8; ++r) {
                    int gm = rowBase + mWave + mt * 16 + r + 8 * half;
                    int gn = nBase + nWave + nt * 16 + (lane & 15);
                    float v = acc[mt][nt][r];
                    Wh1f[gm * DM + gn] = v;
                    Wh1b[gm * DM + gn] = f32_to_bf16(v);
                }
    }
}

// ===========================================================================
// K2: score vectors (wave per row, coalesced, wave32 shfl reduction)
// ===========================================================================
__global__ __launch_bounds__(256)
void k2_scores(const float* __restrict__ Wh0f, const float* __restrict__ Wh1f,
               const float* __restrict__ va,  const float* __restrict__ v1,
               const float* __restrict__ v2,  const float* __restrict__ v3,
               float* __restrict__ svec)
{
    const int lane = threadIdx.x & 31;
    const int row  = blockIdx.x * 8 + (threadIdx.x >> 5);
    if (row >= BNTOT) return;
    const float* vecs[4] = { va, v1, v2, v3 };

    float accLo[4] = {0,0,0,0}, accHi[4] = {0,0,0,0};
    for (int d = lane; d < DM; d += 32) {
        float w0 = Wh0f[row * DM + d];
        float w1 = Wh1f[row * DM + d];
        #pragma unroll
        for (int v = 0; v < 4; ++v) {
            accLo[v] = fmaf(w0, vecs[v][d],      accLo[v]);
            accHi[v] = fmaf(w1, vecs[v][DM + d], accHi[v]);
        }
    }
    #pragma unroll
    for (int v = 0; v < 4; ++v)
        for (int off = 16; off > 0; off >>= 1) {
            accLo[v] += __shfl_down(accLo[v], off, 32);
            accHi[v] += __shfl_down(accHi[v], off, 32);
        }
    if (lane == 0) {
        #pragma unroll
        for (int v = 0; v < 4; ++v) {
            svec[(v * 2 + 0) * BNTOT + row] = accLo[v];
            svec[(v * 2 + 1) * BNTOT + row] = accHi[v];
        }
    }
}

// ===========================================================================
// K3: online-softmax column stats over axis i (softmax axis=1), 8-way i-split
// ===========================================================================
__global__ __launch_bounds__(256)
void k3_colstats(const float* __restrict__ svec,
                 const int* __restrict__ e1, const int* __restrict__ e2,
                 const int* __restrict__ e3,
                 float* __restrict__ pm, float* __restrict__ ps)
{
    const int b  = blockIdx.x;
    const int j  = blockIdx.y * 256 + threadIdx.x;
    const int i0 = blockIdx.z * (NN / 8);

    const float saj = svec[1 * BNTOT + b * NN + j];
    const float s1j = svec[3 * BNTOT + b * NN + j];
    const float s2j = svec[5 * BNTOT + b * NN + j];
    const float s3j = svec[7 * BNTOT + b * NN + j];

    float m = -3.4e38f, s = 0.0f;
    const int ebase = b * NN * NN + j;
    for (int i = i0; i < i0 + NN / 8; ++i) {
        float sai = svec[0 * BNTOT + b * NN + i];
        float s1i = svec[2 * BNTOT + b * NN + i];
        float s2i = svec[4 * BNTOT + b * NN + i];
        float s3i = svec[6 * BNTOT + b * NN + i];
        int eo = ebase + i * NN;
        float t1 = (e1[eo] > 0) ? lrelu(s1i + s1j) : 0.0f;
        float t2 = (e2[eo] > 0) ? lrelu(s2i + s2j) : 0.0f;
        float t3 = (e3[eo] > 0) ? lrelu(s3i + s3j) : 0.0f;
        float L = lrelu(sai + saj) + 0.01f * (t1 + t2 + t3);
        float nm = fmaxf(m, L);
        s = s * __expf(m - nm) + __expf(L - nm);
        m = nm;
    }
    pm[blockIdx.z * BNTOT + b * NN + j] = m;
    ps[blockIdx.z * BNTOT + b * NN + j] = s;
}

__global__ __launch_bounds__(256)
void k3b_combine(const float* __restrict__ pm, const float* __restrict__ ps,
                 float* __restrict__ colmax, float* __restrict__ colrcp)
{
    const int idx = blockIdx.x * 256 + threadIdx.x;
    if (idx >= BNTOT) return;
    float M = -3.4e38f;
    #pragma unroll
    for (int k = 0; k < 8; ++k) M = fmaxf(M, pm[k * BNTOT + idx]);
    float S = 0.0f;
    #pragma unroll
    for (int k = 0; k < 8; ++k) S += ps[k * BNTOT + idx] * __expf(pm[k * BNTOT + idx] - M);
    colmax[idx] = M;
    colrcp[idx] = 1.0f / S;
}

// ===========================================================================
// K4: h_prime = attention @ Wh1 (+elu), fused.
// grid (N/16, B), 256 threads (8 waves).  Per j-step (32 wide):
//  - TDM double-buffered load of the 32x768 Wh1(bf16) slab into LDS (wave 0,
//    s_wait_tensorcnt pipelining)
//  - all threads build the 16x32 attention tile in LDS (bf16 A operand)
//  - 8 waves x 6 d-tiles of v_wmma_f32_16x16x32_bf16
// ===========================================================================
__global__ __launch_bounds__(256)
void k4_agg(const float* __restrict__ svec,
            const int* __restrict__ e1, const int* __restrict__ e2,
            const int* __restrict__ e3,
            const float* __restrict__ colmax, const float* __restrict__ colrcp,
            const unsigned short* __restrict__ Wh1b,
            float* __restrict__ out)
{
    __shared__ unsigned short BLds[2][32 * DM];   // double-buffered slab, [k][d]
    __shared__ unsigned short attLds[16][34];     // A operand (i x k, bf16)

    const int t     = threadIdx.x;
    const int lane  = t & 31;
    const int half  = lane >> 4;
    const int wave  = t >> 5;
    const int i0    = blockIdx.x * 16;
    const int b     = blockIdx.y;
    const int dBase = wave * 96;
    const int STEPS = NN / 32;

    v8f acc[6];
    #pragma unroll
    for (int nt = 0; nt < 6; ++nt) acc[nt] = (v8f)(0.0f);

    // prologue: TDM load of slab 0 into buffer 0
    if (wave == 0) {
        tdm_load_2d_b16((unsigned)(size_t)&BLds[0][0],
                        Wh1b + (size_t)(b * NN) * DM,
                        DM, NN, DM, 32, DM);
    }

    for (int s = 0; s < STEPS; ++s) {
        const int j0  = s * 32;
        const int cur = s & 1;
        __syncthreads();   // both buffers free for (re)use

        // ---- attention tile 16x32 (2 elements / thread), coalesced along j
        #pragma unroll
        for (int q = 0; q < 2; ++q) {
            int idx = t + 256 * q;             // 0..511
            int k = idx & 31, i = idx >> 5;
            int gi = i0 + i, j = j0 + k;
            float sai = svec[0 * BNTOT + b * NN + gi];
            float s1i = svec[2 * BNTOT + b * NN + gi];
            float s2i = svec[4 * BNTOT + b * NN + gi];
            float s3i = svec[6 * BNTOT + b * NN + gi];
            float saj = svec[1 * BNTOT + b * NN + j];
            float s1j = svec[3 * BNTOT + b * NN + j];
            float s2j = svec[5 * BNTOT + b * NN + j];
            float s3j = svec[7 * BNTOT + b * NN + j];
            int eo = (b * NN + gi) * NN + j;
            if (q == 0 && j0 + 32 < NN) {      // pull next mask lines toward L2
                __builtin_prefetch(&e1[eo + 32], 0, 1);
                __builtin_prefetch(&e2[eo + 32], 0, 1);
                __builtin_prefetch(&e3[eo + 32], 0, 1);
            }
            float t1 = (e1[eo] > 0) ? lrelu(s1i + s1j) : 0.0f;
            float t2 = (e2[eo] > 0) ? lrelu(s2i + s2j) : 0.0f;
            float t3 = (e3[eo] > 0) ? lrelu(s3i + s3j) : 0.0f;
            float L = lrelu(sai + saj) + 0.01f * (t1 + t2 + t3);
            float att = __expf(L - colmax[b * NN + j]) * colrcp[b * NN + j];
            attLds[i][k] = f32_to_bf16(att);
        }

        // ---- TDM: launch next slab, then guarantee current slab landed
        if (wave == 0) {
            if (s + 1 < STEPS) {
                tdm_load_2d_b16((unsigned)(size_t)&BLds[1 - cur][0],
                                Wh1b + (size_t)(b * NN + j0 + 32) * DM,
                                DM, NN, DM, 32, DM);
                __builtin_amdgcn_s_wait_tensorcnt(1);  // current slab done
            } else {
                __builtin_amdgcn_s_wait_tensorcnt(0);  // last slab done
            }
        }
        __syncthreads();   // slab + attention tile visible to all waves

        // ---- fragments + WMMA
        v8u rawA;
        #pragma unroll
        for (int v = 0; v < 8; ++v)
            rawA[v] = *(const unsigned int*)&attLds[lane & 15][kk_a(v, half)];
        v16bf aF = __builtin_bit_cast(v16bf, rawA);

        #pragma unroll
        for (int nt = 0; nt < 6; ++nt) {
            int d = dBase + nt * 16 + (lane & 15);
            v8u rawB;
            #pragma unroll
            for (int v = 0; v < 8; ++v) {
                int k = kk_b(v, half);
                unsigned lo = BLds[cur][k * DM + d];
                unsigned hi = BLds[cur][(k + 1) * DM + d];
                rawB[v] = lo | (hi << 16);
            }
            v16bf bF = __builtin_bit_cast(v16bf, rawB);
            acc[nt] = __builtin_amdgcn_wmma_f32_16x16x32_bf16(
                false, aF, false, bF, (short)0, acc[nt], false, false);
        }
    }

    // ---- epilogue: elu, write f32
    #pragma unroll
    for (int nt = 0; nt < 6; ++nt)
        #pragma unroll
        for (int r = 0; r < 8; ++r) {
            int gi = i0 + r + 8 * half;
            int gd = dBase + nt * 16 + (lane & 15);
            float x = acc[nt][r];
            out[(b * NN + gi) * DM + gd] = x > 0.0f ? x : (__expf(x) - 1.0f);
        }
}

// ===========================================================================
extern "C" void kernel_launch(void* const* d_in, const int* in_sizes, int n_in,
                              void* d_out, int out_size, void* d_ws, size_t ws_size,
                              hipStream_t stream)
{
    const float* h0 = (const float*)d_in[0];
    const float* h1 = (const float*)d_in[1];
    const float* n1 = (const float*)d_in[2];
    const int*   e1 = (const int*)  d_in[3];
    const float* n2 = (const float*)d_in[4];
    const int*   e2 = (const int*)  d_in[5];
    const float* n3 = (const float*)d_in[6];
    const int*   e3 = (const int*)  d_in[7];
    const float* W  = (const float*)d_in[8];
    const float* va = (const float*)d_in[9];
    float* out = (float*)d_out;

    // workspace layout (~64 MB)
    char* ws = (char*)d_ws;
    float*          Wh0f   = (float*)ws;
    float*          Wh1f   = Wh0f + (size_t)BNTOT * DM;
    unsigned short* Wh1b   = (unsigned short*)(Wh1f + (size_t)BNTOT * DM);
    float*          svec   = (float*)(Wh1b + (size_t)BNTOT * DM);
    float*          pm     = svec + 8 * BNTOT;
    float*          ps     = pm   + 8 * BNTOT;
    float*          colmax = ps   + 8 * BNTOT;
    float*          colrcp = colmax + BNTOT;

    k1_proj<<<dim3(BNTOT / 128, DM / 128, 2), 256, 0, stream>>>(
        h0, h1, W, Wh0f, Wh1f, Wh1b);
    k2_scores<<<BNTOT / 8, 256, 0, stream>>>(Wh0f, Wh1f, va, n1, n2, n3, svec);
    k3_colstats<<<dim3(BATCH, NN / 256, 8), 256, 0, stream>>>(
        svec, e1, e2, e3, pm, ps);
    k3b_combine<<<BNTOT / 256, 256, 0, stream>>>(pm, ps, colmax, colrcp);
    k4_agg<<<dim3(NN / 16, BATCH), 256, 0, stream>>>(
        svec, e1, e2, e3, colmax, colrcp, Wh1b, out);
}
